// AttentivePooling_20272245637420
// MI455X (gfx1250) — matrix operand order repaired and checked by reading, not compiled, not run on hardware
//
#include <hip/hip_runtime.h>
#include <hip/hip_bf16.h>

// ---------------------------------------------------------------------------
// AttentivePooling on MI455X (gfx1250, wave32, WMMA)
//   pass 0: x, W1 -> bf16 (streaming convert, done once)
//   pass 1: h = tanh(x @ W1^T + b1)  bf16 WMMA GEMM, f32 acc, async LDS copies
//   pass 2: q / scores / softmax / combine (memory-streaming over bf16 h)
// ---------------------------------------------------------------------------

#define SEQ   250
#define BATCH 512
#define HID   1024
#define MTOT  (SEQ * BATCH)   // 128000 rows, divisible by 128
#define BM    128
#define BN    128
#define BK    64              // k-tile (bf16 elements)
#define LDSW  36              // padded dwords per 64-bf16 row (bank-conflict free)

typedef __attribute__((ext_vector_type(16))) __bf16   bf16x16;
typedef __attribute__((ext_vector_type(8)))  float    f32x8;
typedef __attribute__((ext_vector_type(4)))  unsigned u32x4;
typedef int v4i __attribute__((__vector_size__(16)));           // int4 vector
typedef __attribute__((address_space(1))) v4i* gptr_v4i;        // global int4*
typedef __attribute__((address_space(3))) v4i* lptr_v4i;        // LDS    int4*

#if defined(__has_builtin)
#if __has_builtin(__builtin_amdgcn_global_load_async_to_lds_b128)
#define HAVE_ASYNC_LDS 1
#endif
#endif
#ifndef HAVE_ASYNC_LDS
#define HAVE_ASYNC_LDS 0
#endif

__device__ __forceinline__ unsigned pack_bf16x2(float lo, float hi) {
  unsigned ul = __builtin_bit_cast(unsigned, lo);
  unsigned uh = __builtin_bit_cast(unsigned, hi);
  ul += 0x7FFFu + ((ul >> 16) & 1u);            // round-to-nearest-even
  uh += 0x7FFFu + ((uh >> 16) & 1u);
  return (ul >> 16) | (uh & 0xFFFF0000u);
}

__device__ __forceinline__ unsigned short f2bf_bits(float f) {
  unsigned u = __builtin_bit_cast(unsigned, f);
  u += 0x7FFFu + ((u >> 16) & 1u);
  return (unsigned short)(u >> 16);
}

__device__ __forceinline__ float bf_lo(unsigned p) {
  return __builtin_bit_cast(float, p << 16);
}
__device__ __forceinline__ float bf_hi(unsigned p) {
  return __builtin_bit_cast(float, p & 0xFFFF0000u);
}

// ---------------------------------------------------------------------------
// Pass 0: fp32 -> packed bf16 (2/dword), one float4 -> uint2 per thread.
// ---------------------------------------------------------------------------
__global__ __launch_bounds__(256) void ap_cvt_bf16(
    const float4* __restrict__ src, uint2* __restrict__ dst, int n4)
{
  const int i = blockIdx.x * 256 + threadIdx.x;
  if (i < n4) {
    const float4 v = src[i];
    dst[i] = make_uint2(pack_bf16x2(v.x, v.y), pack_bf16x2(v.z, v.w));
  }
}

// ---------------------------------------------------------------------------
// Pass 1 (fast path): bf16 GEMM + tanh.  128x128 tile, 256 threads (8 waves),
// K-tile 64, double-buffered LDS filled by async global->LDS b128 copies.
// ---------------------------------------------------------------------------
__global__ __launch_bounds__(256) void ap_gemm_tanh_bf16(
    const unsigned* __restrict__ xb,   // [MTOT, HID/2] packed bf16
    const unsigned* __restrict__ wb,   // [HID,  HID/2] packed bf16
    const float* __restrict__ b1,      // [HID]
    unsigned short* __restrict__ hb)   // [MTOT, HID] bf16
{
  __shared__ unsigned As[2][BM * LDSW];   // 2 x 18 KB
  __shared__ unsigned Bs[2][BM * LDSW];

  const int t     = threadIdx.x;
  const int lane  = t & 31;
  const int w     = t >> 5;
  const int waveM = w >> 2;          // 0..1  -> 64 rows
  const int waveN = w & 3;           // 0..3  -> 32 cols
  const int hh    = lane >> 4;
  const int lr    = lane & 15;

  const long rowBase = (long)blockIdx.y * BM;
  const int  colBase = blockIdx.x * BN;

  const unsigned* xg = xb + rowBase * (HID / 2);
  const unsigned* wg = wb + (long)colBase * (HID / 2);

  f32x8 zero = {0.f, 0.f, 0.f, 0.f, 0.f, 0.f, 0.f, 0.f};
  f32x8 acc[4][2];
#pragma unroll
  for (int i = 0; i < 4; ++i)
#pragma unroll
    for (int j = 0; j < 2; ++j) acc[i][j] = zero;

  // Per-stage copy: 128 rows x 8 16B-chunks each for A and B; 4+4 per thread.
#if HAVE_ASYNC_LDS
  auto issue = [&](int kt, int buf) {
#pragma unroll
    for (int j = 0; j < 4; ++j) {
      const int i    = t + 256 * j;
      const int row  = i >> 3;
      const int c    = i & 7;
      const int ldso = row * LDSW + c * 4;                    // dwords
      const long go  = (long)row * (HID / 2) + kt * (BK / 2) + c * 4;
      __builtin_amdgcn_global_load_async_to_lds_b128(
          (gptr_v4i)(v4i*)(xg + go), (lptr_v4i)(v4i*)&As[buf][ldso], 0, 0);
      __builtin_amdgcn_global_load_async_to_lds_b128(
          (gptr_v4i)(v4i*)(wg + go), (lptr_v4i)(v4i*)&Bs[buf][ldso], 0, 0);
    }
  };
#else
  uint4 va[4], vb[4];
  auto fetch = [&](int kt) {
#pragma unroll
    for (int j = 0; j < 4; ++j) {
      const int i   = t + 256 * j;
      const int row = i >> 3;
      const int c   = i & 7;
      const long go = (long)row * (HID / 2) + kt * (BK / 2) + c * 4;
      va[j] = *(const uint4*)(xg + go);
      vb[j] = *(const uint4*)(wg + go);
    }
  };
  auto stash = [&](int buf) {
#pragma unroll
    for (int j = 0; j < 4; ++j) {
      const int i    = t + 256 * j;
      const int row  = i >> 3;
      const int c    = i & 7;
      const int ldso = row * LDSW + c * 4;
      *(uint4*)&As[buf][ldso] = va[j];
      *(uint4*)&Bs[buf][ldso] = vb[j];
    }
  };
#endif

  auto compute = [&](int buf) {
#pragma unroll
    for (int ks = 0; ks < 2; ++ks) {          // two K=32 sub-steps per stage
      const int kd = ks * 16;                 // dword offset of sub-step
      bf16x16 fb[2];
#pragma unroll
      for (int nt = 0; nt < 2; ++nt) {        // B: lane holds N=nt*16+lr,
        const int r = waveN * 32 + nt * 16 + lr;  // K = hh*16..hh*16+15
        const unsigned* base = &Bs[buf][r * LDSW + kd + hh * 8];
        union { bf16x16 v; u32x4 q[2]; } u;
        u.q[0] = *(const u32x4*)(base);
        u.q[1] = *(const u32x4*)(base + 4);
        fb[nt] = u.v;
      }
#pragma unroll
      for (int mt = 0; mt < 4; ++mt) {        // A: K = hh*16..hh*16+15
        const int r = waveM * 64 + mt * 16 + lr;
        const unsigned* base = &As[buf][r * LDSW + kd];
        union { bf16x16 v; u32x4 q[2]; } u;
        u.q[0] = *(const u32x4*)(base + hh * 8);
        u.q[1] = *(const u32x4*)(base + hh * 8 + 4);
        const bf16x16 fa = u.v;
#pragma unroll
        for (int nt = 0; nt < 2; ++nt)
          acc[mt][nt] = __builtin_amdgcn_wmma_f32_16x16x32_bf16(
              false, fa, false, fb[nt], (short)0, acc[mt][nt], false, false);
      }
    }
  };

  const int KT = HID / BK;  // 16 stages
#if HAVE_ASYNC_LDS
  issue(0, 0);
  for (int kt = 0; kt < KT; ++kt) {
    if (kt + 1 < KT) {
      issue(kt + 1, (kt + 1) & 1);
      asm volatile("s_wait_asynccnt 0x8" ::: "memory");  // stage kt landed
    } else {
      asm volatile("s_wait_asynccnt 0x0" ::: "memory");
    }
    __syncthreads();          // all waves' copies visible
    compute(kt & 1);
    __syncthreads();          // safe to overwrite buf (kt&1) next iteration
  }
#else
  fetch(0);
  stash(0);
  __syncthreads();
  for (int kt = 0; kt < KT; ++kt) {
    if (kt + 1 < KT) fetch(kt + 1);
    compute(kt & 1);
    __syncthreads();
    if (kt + 1 < KT) {
      stash((kt + 1) & 1);
      __syncthreads();
    }
  }
#endif

  // Epilogue: D layout -> M = hh*8 + v, N = lr per 16x16 tile
#pragma unroll
  for (int nt = 0; nt < 2; ++nt) {
    const int col  = colBase + waveN * 32 + nt * 16 + lr;
    const float bc = b1[col];
#pragma unroll
    for (int mt = 0; mt < 4; ++mt) {
      const long row0 = rowBase + waveM * 64 + mt * 16 + hh * 8;
#pragma unroll
      for (int v = 0; v < 8; ++v) {
        const float o = tanhf(acc[mt][nt][v] + bc);
        hb[(row0 + v) * HID + col] = f2bf_bits(o);
      }
    }
  }
}

// ---------------------------------------------------------------------------
// Pass 1 (fallback when ws is small): fp32 inputs, convert-in-kernel, BK=32.
// ---------------------------------------------------------------------------
__global__ __launch_bounds__(256) void ap_gemm_tanh_direct(
    const float* __restrict__ x, const float* __restrict__ W1,
    const float* __restrict__ b1, unsigned short* __restrict__ hb)
{
  __shared__ unsigned As[2][BM * 16];
  __shared__ unsigned Bs[2][BM * 16];

  const int t     = threadIdx.x;
  const int lane  = t & 31;
  const int w     = t >> 5;
  const int waveM = w >> 2;
  const int waveN = w & 3;
  const int hh    = lane >> 4;
  const int lr    = lane & 15;

  const long rowBase = (long)blockIdx.y * BM;
  const int  colBase = blockIdx.x * BN;
  const float* xg = x + rowBase * HID;
  const float* wg = W1 + (long)colBase * HID;

  f32x8 zero = {0.f, 0.f, 0.f, 0.f, 0.f, 0.f, 0.f, 0.f};
  f32x8 acc[4][2];
#pragma unroll
  for (int i = 0; i < 4; ++i)
#pragma unroll
    for (int j = 0; j < 2; ++j) acc[i][j] = zero;

  float4 va[4], vb[4];
  auto fetch = [&](int kt) {
    const int k0 = kt * 32;
#pragma unroll
    for (int j = 0; j < 4; ++j) {
      const int i = t + 256 * j, row = i >> 3, c = i & 7;
      va[j] = *(const float4*)(xg + (long)row * HID + k0 + c * 4);
      vb[j] = *(const float4*)(wg + (long)row * HID + k0 + c * 4);
    }
  };
  auto stash = [&](int buf) {
#pragma unroll
    for (int j = 0; j < 4; ++j) {
      const int i = t + 256 * j, row = i >> 3, c = i & 7, off = row * 16 + c * 2;
      *(uint2*)&As[buf][off] =
          make_uint2(pack_bf16x2(va[j].x, va[j].y), pack_bf16x2(va[j].z, va[j].w));
      *(uint2*)&Bs[buf][off] =
          make_uint2(pack_bf16x2(vb[j].x, vb[j].y), pack_bf16x2(vb[j].z, vb[j].w));
    }
  };
  auto compute = [&](int buf) {
    bf16x16 fb[2];
#pragma unroll
    for (int nt = 0; nt < 2; ++nt) {
      const int r = waveN * 32 + nt * 16 + lr;
      const unsigned* base = &Bs[buf][r * 16 + hh * 8];
      union { bf16x16 v; u32x4 q[2]; } u;
      u.q[0] = *(const u32x4*)(base);
      u.q[1] = *(const u32x4*)(base + 4);
      fb[nt] = u.v;
    }
#pragma unroll
    for (int mt = 0; mt < 4; ++mt) {
      const int r = waveM * 64 + mt * 16 + lr;
      const unsigned* base = &As[buf][r * 16];
      union { bf16x16 v; u32x4 q[2]; } u;
      u.q[0] = *(const u32x4*)(base + hh * 8);
      u.q[1] = *(const u32x4*)(base + hh * 8 + 4);
      const bf16x16 fa = u.v;
#pragma unroll
      for (int nt = 0; nt < 2; ++nt)
        acc[mt][nt] = __builtin_amdgcn_wmma_f32_16x16x32_bf16(
            false, fa, false, fb[nt], (short)0, acc[mt][nt], false, false);
    }
  };

  fetch(0); stash(0); __syncthreads();
  const int KT = HID / 32;
  for (int kt = 0; kt < KT; ++kt) {
    if (kt + 1 < KT) fetch(kt + 1);
    compute(kt & 1);
    __syncthreads();
    if (kt + 1 < KT) { stash((kt + 1) & 1); __syncthreads(); }
  }

#pragma unroll
  for (int nt = 0; nt < 2; ++nt) {
    const int col  = colBase + waveN * 32 + nt * 16 + lr;
    const float bc = b1[col];
#pragma unroll
    for (int mt = 0; mt < 4; ++mt) {
      const long row0 = rowBase + waveM * 64 + mt * 16 + hh * 8;
#pragma unroll
      for (int v = 0; v < 8; ++v)
        hb[(row0 + v) * HID + col] = f2bf_bits(tanhf(acc[mt][nt][v] + bc));
    }
  }
}

// ---------------------------------------------------------------------------
// Pass 2: per-batch attention. One block (256 threads, 8 waves) per b.
// ---------------------------------------------------------------------------
__global__ __launch_bounds__(256) void ap_attend(
    const unsigned* __restrict__ hbq,  // [SEQ, BATCH, HID/2] packed bf16
    const float* __restrict__ W2, const float* __restrict__ b2,
    float* __restrict__ out)           // [BATCH, HID]
{
  __shared__ float qs[HID];
  __shared__ float sc[256];
  __shared__ float w2s[256];

  const int b    = blockIdx.x;
  const int t    = threadIdx.x;
  const int lane = t & 31;
  const int w    = t >> 5;
  const int SU   = BATCH * (HID / 2);
  const unsigned* hbB = hbq + (size_t)b * (HID / 2);

  if (t < SEQ) w2s[t] = W2[t];
  __syncthreads();

  const float bb = b2[0];
#pragma unroll
  for (int j = 0; j < 2; ++j) {        // q
    const int cu = t + j * 256;
    float a0 = bb, a1 = bb;
    for (int s = 0; s < SEQ; ++s) {
      const unsigned p = hbB[(size_t)s * SU + cu];
      a0 = fmaf(w2s[s], bf_lo(p), a0);
      a1 = fmaf(w2s[s], bf_hi(p), a1);
    }
    qs[2 * cu] = a0;
    qs[2 * cu + 1] = a1;
  }
  __syncthreads();

  for (int s = w; s < SEQ; s += 8) {   // scores
    const unsigned* hr = hbB + (size_t)s * SU;
    float a = 0.f;
    for (int cu = lane; cu < HID / 2; cu += 32) {
      const unsigned p = hr[cu];
      a = fmaf(bf_lo(p), qs[2 * cu], a);
      a = fmaf(bf_hi(p), qs[2 * cu + 1], a);
    }
#pragma unroll
    for (int off = 16; off > 0; off >>= 1) a += __shfl_down(a, off, 32);
    if (lane == 0) sc[s] = a;
  }
  __syncthreads();

  if (w == 0) {                        // softmax over s
    float m = -1e30f;
    for (int s = lane; s < SEQ; s += 32) m = fmaxf(m, sc[s]);
#pragma unroll
    for (int off = 16; off > 0; off >>= 1) m = fmaxf(m, __shfl_xor(m, off, 32));
    float sum = 0.f;
    for (int s = lane; s < SEQ; s += 32) {
      const float e = __expf(sc[s] - m);
      sc[s] = e; sum += e;
    }
#pragma unroll
    for (int off = 16; off > 0; off >>= 1) sum += __shfl_xor(sum, off, 32);
    const float inv = 1.f / sum;
    for (int s = lane; s < SEQ; s += 32) sc[s] *= inv;
  }
  __syncthreads();

#pragma unroll
  for (int j = 0; j < 2; ++j) {        // combine
    const int cu = t + j * 256;
    float a0 = 0.f, a1 = 0.f;
    for (int s = 0; s < SEQ; ++s) {
      const unsigned p = hbB[(size_t)s * SU + cu];
      a0 = fmaf(sc[s], bf_lo(p), a0);
      a1 = fmaf(sc[s], bf_hi(p), a1);
    }
    *(float2*)&out[(size_t)b * HID + 2 * cu] = make_float2(a0, a1);
  }
}

// ---------------------------------------------------------------------------
extern "C" void kernel_launch(void* const* d_in, const int* in_sizes, int n_in,
                              void* d_out, int out_size, void* d_ws, size_t ws_size,
                              hipStream_t stream) {
  (void)in_sizes; (void)n_in; (void)out_size;

  const float* x  = (const float*)d_in[0];   // [SEQ, BATCH, HID]
  const float* W1 = (const float*)d_in[1];   // [HID, HID]
  const float* b1 = (const float*)d_in[2];   // [HID]
  const float* W2 = (const float*)d_in[3];   // [1, SEQ]
  const float* b2 = (const float*)d_in[4];   // [1]
  float* out = (float*)d_out;                // [BATCH, HID]

  const size_t hbytes = (size_t)MTOT * HID * 2;  // 256 MB bf16 h
  const size_t xbytes = (size_t)MTOT * HID * 2;  // 256 MB bf16 x
  const size_t wbytes = (size_t)HID * HID * 2;   //   2 MB bf16 W1

  unsigned short* hbuf = (unsigned short*)d_ws;
  dim3 g1(HID / BN, MTOT / BM);  // (8, 1000): N fastest for L2 reuse of x rows

  if (ws_size >= hbytes + xbytes + wbytes) {
    unsigned* xb = (unsigned*)((char*)d_ws + hbytes);
    unsigned* wb = (unsigned*)((char*)d_ws + hbytes + xbytes);
    const int n4x = MTOT * HID / 4;   // 32,768,000 -> 128000 blocks
    const int n4w = HID * HID / 4;    //    262,144 ->   1024 blocks
    ap_cvt_bf16<<<n4x / 256, 256, 0, stream>>>((const float4*)x, (uint2*)xb, n4x);
    ap_cvt_bf16<<<n4w / 256, 256, 0, stream>>>((const float4*)W1, (uint2*)wb, n4w);
    ap_gemm_tanh_bf16<<<g1, 256, 0, stream>>>(xb, wb, b1, hbuf);
  } else {
    ap_gemm_tanh_direct<<<g1, 256, 0, stream>>>(x, W1, b1, hbuf);
  }

  ap_attend<<<BATCH, 256, 0, stream>>>((const unsigned*)hbuf, W2, b2, out);
}
